// WaveletEncoder_30829275251251
// MI455X (gfx1250) — compile-verified, matrix operand order checked by reading, not run on hardware
//
#include <hip/hip_runtime.h>
#include <hip/hip_bf16.h>

typedef __attribute__((ext_vector_type(4)))  float    v4f;
typedef __attribute__((ext_vector_type(16))) _Float16 v16h;
typedef __attribute__((ext_vector_type(8)))  float    v8f;

#define NB  16      // batch
#define NC  256     // channels
#define HW  128
#define PLANE (HW*HW)         // 16384 floats per (b,c) plane
#define NPLANE (NB*NC)        // 4096
#define AGGW 1536             // 6*C
#define H1  256               // hidden width (2*DF)
#define DF  128

// ---------------------------------------------------------------------------
// Kernel 1: streaming reduction of x -> 16 mod-4 phase sums per (b,c) plane.
// One block per plane. 256 threads, 16 x float4 NT loads each (fully coalesced
// b128). Row-phase per thread is constant = (tid>>5)&3 (stride 256 float4s ==
// 8 rows), so each wave holds a single row-phase -> wave32 shfl_xor reduce.
// ---------------------------------------------------------------------------
__global__ __launch_bounds__(256) void phase_sum_kernel(
    const float* __restrict__ x, float* __restrict__ S) {
  const int plane = blockIdx.x;
  const v4f* p = (const v4f*)(x + (size_t)plane * PLANE);
  const int t = threadIdx.x;

  float a0 = 0.f, a1 = 0.f, a2 = 0.f, a3 = 0.f;   // col phases 0..3
#pragma unroll
  for (int k = 0; k < 16; ++k) {
    v4f v = __builtin_nontemporal_load(&p[t + 256 * k]);
    a0 += v.x; a1 += v.y; a2 += v.z; a3 += v.w;
  }
  // wave32 reduction (all lanes of a wave share the same row phase)
#pragma unroll
  for (int m = 16; m >= 1; m >>= 1) {
    a0 += __shfl_xor(a0, m, 32);
    a1 += __shfl_xor(a1, m, 32);
    a2 += __shfl_xor(a2, m, 32);
    a3 += __shfl_xor(a3, m, 32);
  }
  __shared__ float red[8][4];
  if ((t & 31) == 0) {
    const int w = t >> 5;
    red[w][0] = a0; red[w][1] = a1; red[w][2] = a2; red[w][3] = a3;
  }
  __syncthreads();
  if (t < 16) {
    const int rp = t >> 2, cp = t & 3;            // waves rp and rp+4 share row phase
    S[(size_t)plane * 16 + t] = red[rp][cp] + red[rp + 4][cp];
  }
}

// ---------------------------------------------------------------------------
// Kernel 2: fold phase sums with the per-channel 2x2 filters into agg(16,1536).
// Detail order per reference: [lh0, hl0, hh0, lh1, hl1, hh1], each (B,C).
// ---------------------------------------------------------------------------
__global__ __launch_bounds__(256) void combine_kernel(
    const float* __restrict__ S,
    const float* __restrict__ wll0, const float* __restrict__ wlh0,
    const float* __restrict__ whl0, const float* __restrict__ whh0,
    const float* __restrict__ wlh1, const float* __restrict__ whl1,
    const float* __restrict__ whh1,
    float* __restrict__ agg) {
  const int idx = blockIdx.x * blockDim.x + threadIdx.x;   // 0..4095
  if (idx >= NPLANE) return;
  const int b = idx >> 8, c = idx & 255;

  float s[4][4];
#pragma unroll
  for (int r = 0; r < 4; ++r)
#pragma unroll
    for (int cc = 0; cc < 4; ++cc)
      s[r][cc] = S[(size_t)idx * 16 + r * 4 + cc];

  // scale 0: mod-2 phase sums; band mean = (1/4096) * sum_pq w[p][q]*m2[p][q]
  float m2[2][2];
#pragma unroll
  for (int p = 0; p < 2; ++p)
#pragma unroll
    for (int q = 0; q < 2; ++q)
      m2[p][q] = s[p][q] + s[p][q + 2] + s[p + 2][q] + s[p + 2][q + 2];

  const float inv0 = 1.0f / 4096.0f;
  float d0[3] = {0.f, 0.f, 0.f};
  const float* w0[3] = {wlh0, whl0, whh0};
#pragma unroll
  for (int bnd = 0; bnd < 3; ++bnd) {
    float acc = 0.f;
#pragma unroll
    for (int p = 0; p < 2; ++p)
#pragma unroll
      for (int q = 0; q < 2; ++q)
        acc += w0[bnd][c * 4 + p * 2 + q] * m2[p][q];
    d0[bnd] = acc * inv0;
  }

  // scale 1: e[p1][q1] = sum_{p0,q0} wll0[p0,q0] * s[2p1+p0][2q1+q0]
  float e[2][2];
#pragma unroll
  for (int p1 = 0; p1 < 2; ++p1)
#pragma unroll
    for (int q1 = 0; q1 < 2; ++q1) {
      float acc = 0.f;
#pragma unroll
      for (int p0 = 0; p0 < 2; ++p0)
#pragma unroll
        for (int q0 = 0; q0 < 2; ++q0)
          acc += wll0[c * 4 + p0 * 2 + q0] * s[2 * p1 + p0][2 * q1 + q0];
      e[p1][q1] = acc;
    }

  const float inv1 = 1.0f / 1024.0f;
  float d1[3] = {0.f, 0.f, 0.f};
  const float* w1b[3] = {wlh1, whl1, whh1};
#pragma unroll
  for (int bnd = 0; bnd < 3; ++bnd) {
    float acc = 0.f;
#pragma unroll
    for (int p = 0; p < 2; ++p)
#pragma unroll
      for (int q = 0; q < 2; ++q)
        acc += w1b[bnd][c * 4 + p * 2 + q] * e[p][q];
    d1[bnd] = acc * inv1;
  }

  float* arow = agg + (size_t)b * AGGW;
  arow[0 * NC + c] = d0[0];
  arow[1 * NC + c] = d0[1];
  arow[2 * NC + c] = d0[2];
  arow[3 * NC + c] = d1[0];
  arow[4 * NC + c] = d1[1];
  arow[5 * NC + c] = d1[2];
}

// ---------------------------------------------------------------------------
// Kernel 3: MLP with v_wmma_f32_16x16x32_f16. One block, 16 waves (512 thr).
// Layer 1: wave w computes D tile (16 x 16) at cols w*16 of h = relu(agg@W1.T+b1),
//          K = 1536 -> 48 WMMAs. h stored as f16 in LDS.
// Layer 2: waves 0..7 compute out tiles, K = 256 -> 8 WMMAs.
// Fragment layouts per ISA 7.12.2 (16-bit A 16x32; D: vgpr i -> M=i / M=8+i).
// ---------------------------------------------------------------------------
__global__ __launch_bounds__(512) void mlp_wmma_kernel(
    const float* __restrict__ agg,
    const float* __restrict__ W1, const float* __restrict__ b1,
    const float* __restrict__ W2, const float* __restrict__ b2,
    float* __restrict__ out) {
  __shared__ _Float16 hbuf[16 * H1];   // 16 x 256 f16 = 8 KB

  const int t = threadIdx.x;
  const int lane = t & 31;
  const int wv = t >> 5;
  const int m = lane & 15;
  const int khi = (lane >> 4) << 3;    // lanes 16..31 hold K+8

  // K offset of each of the 16 halves in the A/B fragment (ISA 16-bit layout)
  int kofs[16];
#pragma unroll
  for (int g = 0; g < 8; ++g) {
    const int base = (g < 4) ? (2 * g) : (16 + 2 * (g - 4));
    kofs[2 * g]     = base + khi;
    kofs[2 * g + 1] = base + 1 + khi;
  }

  // ---------------- layer 1 ----------------
  {
    const int n = wv * 16 + m;         // hidden column this lane owns
    v8f acc = {};
    for (int kb = 0; kb < 48; ++kb) {
      const int k0 = kb * 32;
      v16h af, bf;
#pragma unroll
      for (int i = 0; i < 16; ++i) {
        const int k = k0 + kofs[i];
        af[i] = (_Float16)agg[m * AGGW + k];       // A: agg (16 x 1536)
        bf[i] = (_Float16)W1[(size_t)n * AGGW + k];// B[k][n] = W1[n][k]
      }
      acc = __builtin_amdgcn_wmma_f32_16x16x32_f16(
          false, af, false, bf, (short)0, acc, false, false);
    }
    const float bias = b1[n];
#pragma unroll
    for (int i = 0; i < 8; ++i) {
      const int mm = (lane < 16) ? i : (i + 8);    // D layout
      hbuf[mm * H1 + n] = (_Float16)fmaxf(acc[i] + bias, 0.0f);
    }
  }
  __syncthreads();

  // ---------------- layer 2 ---------------- (wave-uniform branch: EXEC all-1s)
  if (wv < 8) {
    const int n = wv * 16 + m;         // output column (0..127)
    v8f acc = {};
#pragma unroll
    for (int kb = 0; kb < 8; ++kb) {
      const int k0 = kb * 32;
      v16h af, bf;
#pragma unroll
      for (int i = 0; i < 16; ++i) {
        const int k = k0 + kofs[i];
        af[i] = hbuf[m * H1 + k];                  // A: h (16 x 256) from LDS
        bf[i] = (_Float16)W2[(size_t)n * H1 + k];  // B[k][n] = W2[n][k]
      }
      acc = __builtin_amdgcn_wmma_f32_16x16x32_f16(
          false, af, false, bf, (short)0, acc, false, false);
    }
    const float bias = b2[n];
#pragma unroll
    for (int i = 0; i < 8; ++i) {
      const int mm = (lane < 16) ? i : (i + 8);
      out[mm * DF + n] = acc[i] + bias;
    }
  }
}

extern "C" void kernel_launch(void* const* d_in, const int* in_sizes, int n_in,
                              void* d_out, int out_size, void* d_ws, size_t ws_size,
                              hipStream_t stream) {
  const float* x    = (const float*)d_in[0];
  const float* wll0 = (const float*)d_in[1];
  const float* wlh0 = (const float*)d_in[2];
  const float* whl0 = (const float*)d_in[3];
  const float* whh0 = (const float*)d_in[4];
  // d_in[5] = w_ll1: unused (LL1 never feeds a detail band)
  const float* wlh1 = (const float*)d_in[6];
  const float* whl1 = (const float*)d_in[7];
  const float* whh1 = (const float*)d_in[8];
  const float* W1   = (const float*)d_in[9];
  const float* b1   = (const float*)d_in[10];
  const float* W2   = (const float*)d_in[11];
  const float* b2   = (const float*)d_in[12];

  float* S   = (float*)d_ws;            // 4096 * 16 floats = 256 KB
  float* agg = S + (size_t)NPLANE * 16; // 16 * 1536 floats =  96 KB

  phase_sum_kernel<<<NPLANE, 256, 0, stream>>>(x, S);
  combine_kernel<<<NPLANE / 256, 256, 0, stream>>>(
      S, wll0, wlh0, whl0, whh0, wlh1, whl1, whh1, agg);
  mlp_wmma_kernel<<<1, 512, 0, stream>>>(agg, W1, b1, W2, b2, (float*)d_out);
}
